// lqformerattention_9792525435200
// MI455X (gfx1250) — matrix operand order repaired and checked by reading, not compiled, not run on hardware
//
#include <hip/hip_runtime.h>
#include <hip/hip_bf16.h>
#include <stdint.h>

// ---------------------------------------------------------------------------
// LQFormer attention for MI455X (gfx1250), bf16 WMMA pipeline.
//   proj_kernel<KDIM,DH,TR> : C = (X @ W^T + b) * outScale (f32 in, bf16 out)
//   attn_kernel : mask stage -> S = q_scaled K^T + mask (mask as WMMA C
//                 operand) -> softmax -> O = P V
// Matmuls lower to v_wmma_f32_16x16x32_bf16 (wave32); the P@V stage streams
// V^T tiles through LDS with the Tensor Data Mover (double-buffered
// tensor_load_to_lds + s_wait_tensorcnt) when the builtin is available.
// ---------------------------------------------------------------------------

typedef __attribute__((ext_vector_type(16))) __bf16 v16bf;
typedef __attribute__((ext_vector_type(8)))  __bf16 v8bf;
typedef __attribute__((ext_vector_type(8)))  float  v8f;
typedef __attribute__((ext_vector_type(4)))  unsigned int v4u;
typedef __attribute__((ext_vector_type(8)))  int    v8i;
typedef __attribute__((ext_vector_type(4)))  int    v4i;

constexpr int kB    = 8;
constexpr int kNH   = 16;
constexpr int kQ    = 1024;
constexpr int kK    = 1024;
constexpr int kE    = 2048;   // EMBED
constexpr int kD    = 512;    // DOWN
constexpr int kDh   = 128;    // EMBED / H
constexpr int kDhd  = 32;     // DOWN / H
constexpr int kSROW = 1040;   // padded LDS row stride (floats) -> bank spread
constexpr int kVROW = 40;     // V-stage LDS row stride in bf16 (80 B, TDM pad)
constexpr int kVBUF = 128 * kVROW;       // one V-stage buffer (bf16 elements)
constexpr unsigned kVBYTES = kVBUF * 2;  // = 10240 bytes

#ifndef __has_builtin
#define __has_builtin(x) 0
#endif
#if __has_builtin(__builtin_amdgcn_tensor_load_to_lds) && \
    __has_builtin(__builtin_amdgcn_s_wait_tensorcnt)
#define HAVE_TDM 1
#else
#define HAVE_TDM 0
#endif

__device__ inline float4 ld4(const float* p) { return *(const float4*)p; }

// Pack 16 consecutive-K f32 values (two groups of 8) into a bf16 WMMA fragment.
__device__ inline v16bf cvt16(float4 a0, float4 a1, float4 b0, float4 b1) {
  v16bf r;
  r[0]  = (__bf16)a0.x; r[1]  = (__bf16)a0.y; r[2]  = (__bf16)a0.z; r[3]  = (__bf16)a0.w;
  r[4]  = (__bf16)a1.x; r[5]  = (__bf16)a1.y; r[6]  = (__bf16)a1.z; r[7]  = (__bf16)a1.w;
  r[8]  = (__bf16)b0.x; r[9]  = (__bf16)b0.y; r[10] = (__bf16)b0.z; r[11] = (__bf16)b0.w;
  r[12] = (__bf16)b1.x; r[13] = (__bf16)b1.y; r[14] = (__bf16)b1.z; r[15] = (__bf16)b1.w;
  return r;
}

__device__ inline v16bf cat8(v8bf lo, v8bf hi) {
  v16bf r;
#pragma unroll
  for (int i = 0; i < 8; i++) { r[i] = lo[i]; r[8 + i] = hi[i]; }
  return r;
}

__device__ inline v8f wmma_bf16(v16bf a, v16bf b, v8f c) {
  return __builtin_amdgcn_wmma_f32_16x16x32_bf16(false, a, false, b, (short)0, c,
                                                 false, false);
}

#if HAVE_TDM
// Issue one TDM load of a 128-row x 32-col bf16 V^T tile into LDS, with
// 4-dword padding per 16 dwords (80 B LDS rows). D# per ISA 08 §8.3/8.4.
__device__ inline void tdm_issue_vtile(const __bf16* vpT, int bh, int kk,
                                       unsigned int lds_off) {
  const uint64_t ga =
      (uint64_t)(uintptr_t)(vpT + (size_t)(bh * kDh) * kK + kk);
  v4u g0 = {1u,                                       // count=1, user mode
            lds_off,                                  // lds_addr (bytes)
            (unsigned int)ga,                         // global_addr[31:0]
            (unsigned int)(ga >> 32) | 0x80000000u};  // [56:32] | type=2
  v8i g1 = {(int)((1u << 16)        // data_size = 2 bytes
                  | (1u << 20)      // pad_enable
                  | (3u << 22)      // pad_interval: 16 dwords
                  | (3u << 25)),    // pad_amount: 4 dwords
            (int)(1024u << 16),     // tensor_dim0 = 1024 (lo16 in hi half)
            (int)(128u << 16),      // tensor_dim0 hi=0 | tensor_dim1 = 128
            (int)(32u << 16),       // tensor_dim1 hi=0 | tile_dim0 = 32
            128,                    // tile_dim1 = 128, tile_dim2 = 0
            1024,                   // tensor_dim0_stride lo32
            0, 0};                  // stride hi, dim1_stride unused (2D)
  v4i gz = {0, 0, 0, 0};
#if __clang_major__ >= 23
  v8i gz8 = {0, 0, 0, 0, 0, 0, 0, 0};
  __builtin_amdgcn_tensor_load_to_lds(g0, g1, gz, gz, gz8, 0);
#else
  __builtin_amdgcn_tensor_load_to_lds(g0, g1, gz, gz, 0);
#endif
}
#endif

// Epilogue: one 16x16 f32 C tile -> (bias add) * outScale -> bf16 store in
// head-split layout. DH is a power of two and L==1024, so div/mod are shifts.
template <int DH, bool TR>
__device__ inline void store_tile(v8f acc, int mt, int nt,
                                  const float* __restrict__ bias,
                                  __bf16* __restrict__ out, int lane,
                                  float outScale) {
  const int l15 = lane & 15, hl = lane >> 4;
  const int col = nt + l15;
  const float bv = bias[col];
  const int h = col / DH, d = col % DH;  // compile-time power-of-two
#pragma unroll
  for (int r = 0; r < 8; r++) {
    const int m  = mt + hl * 8 + r;
    const int bi = m >> 10, l = m & 1023;  // L = 1024
    const float v = (acc[r] + bv) * outScale;
    const int idx = TR ? (((bi * kNH + h) * DH + d) * kQ + l)
                       : (((bi * kNH + h) * kQ + l) * DH + d);
    out[idx] = (__bf16)v;
  }
}

// C[M,N] = (X[M,KDIM] @ W[N,KDIM]^T + bias) * outScale, output bf16 head-split.
// Block: 256 threads = 8 waves, block tile 128x64, wave tile 32x32.
template <int KDIM, int DH, bool TR>
__global__ __launch_bounds__(256) void proj_kernel(
    const float* __restrict__ X, const float* __restrict__ W,
    const float* __restrict__ bias, __bf16* __restrict__ out, float outScale) {
  const int tid = threadIdx.x, lane = tid & 31, wid = tid >> 5;
  const int l15 = lane & 15, hl = lane >> 4;
  const int m0 = blockIdx.y * 128 + (wid >> 1) * 32;
  const int n0 = blockIdx.x * 64 + (wid & 1) * 32;
  const int klo = hl * 8;     // unused: A frag also uses khalf
  const int khalf = hl * 16;  // B-frag K base within 32-step

  v8f z = {};
  v8f acc00 = z, acc01 = z, acc10 = z, acc11 = z;

  for (int kk = 0; kk < KDIM; kk += 32) {
    const float* ar0 = X + (m0 + l15) * KDIM + kk;
    const float* ar1 = X + (m0 + 16 + l15) * KDIM + kk;
    const float* w0 = W + (n0 + l15) * KDIM + kk + khalf;
    const float* w1 = W + (n0 + 16 + l15) * KDIM + kk + khalf;
    __builtin_prefetch(ar0 + 32, 0, 1);
    __builtin_prefetch(w0 + 32, 0, 1);
    v16bf a0 = cvt16(ld4(ar0 + khalf), ld4(ar0 + khalf + 4), ld4(ar0 + khalf + 8),
                     ld4(ar0 + khalf + 12));
    v16bf a1 = cvt16(ld4(ar1 + khalf), ld4(ar1 + khalf + 4), ld4(ar1 + khalf + 8),
                     ld4(ar1 + khalf + 12));
    v16bf b0 = cvt16(ld4(w0), ld4(w0 + 4), ld4(w0 + 8), ld4(w0 + 12));
    v16bf b1 = cvt16(ld4(w1), ld4(w1 + 4), ld4(w1 + 8), ld4(w1 + 12));
    acc00 = wmma_bf16(a0, b0, acc00);
    acc01 = wmma_bf16(a0, b1, acc01);
    acc10 = wmma_bf16(a1, b0, acc10);
    acc11 = wmma_bf16(a1, b1, acc11);
  }
  store_tile<DH, TR>(acc00, m0, n0, bias, out, lane, outScale);
  store_tile<DH, TR>(acc01, m0, n0 + 16, bias, out, lane, outScale);
  store_tile<DH, TR>(acc10, m0 + 16, n0, bias, out, lane, outScale);
  store_tile<DH, TR>(acc11, m0 + 16, n0 + 16, bias, out, lane, outScale);
}

// Fused attention per (b, h, 64 q-rows). 256 threads = 8 waves.
// LDS: S[64][kSROW] f32 (~266 KB) + reduce scratch + 2x10 KB V-stage buffers
// (TDM destinations), all within CDNA5's 320 KB/WGP.
__global__ __launch_bounds__(256) void attn_kernel(
    const __bf16* __restrict__ qp, const __bf16* __restrict__ kp,
    const __bf16* __restrict__ vpT, const float* __restrict__ mask,
    float* __restrict__ outO, float* __restrict__ outP) {
  extern __shared__ float smem[];
  float* S = smem;                          // 64 x kSROW
  float* red = smem + 64 * kSROW;           // 64 x 4 partial reductions
  float* invr = red + 256;                  // 64 per-row 1/sum
  __bf16* vstage = (__bf16*)(invr + 64);    // 2 x (128 x kVROW) bf16

  const int tid = threadIdx.x, lane = tid & 31, wid = tid >> 5;
  const int l15 = lane & 15, hl = lane >> 4;
  const int q0 = blockIdx.x * 64;
  const int h = blockIdx.y, b = blockIdx.z;
  const int bh = b * kNH + h;

#if HAVE_TDM
  const unsigned int vstage_off = (unsigned int)(uintptr_t)(void*)vstage;
  // Kick off the first V^T tile immediately: the DMA overlaps phases 0-2.
  if (wid == 0) tdm_issue_vtile(vpT, bh, 0, vstage_off);
#endif

  // -------- phase 0: stage mask slice into S, coalesced float4 sweep -------
  // (applies the where(mask==0, -1e4, mask) transform on the way in)
  {
    const float* mBase = mask + (size_t)(b * kQ + q0) * kK;
#pragma unroll 2
    for (int j = 0; j < 64; j++) {
      const int idx = (j * 256 + tid) * 4;       // linear over 64x1024
      const int r = idx >> 10, c = idx & 1023;
      float4 mv = *(const float4*)(mBase + idx);
      mv.x = (mv.x == 0.0f) ? -10000.0f : mv.x;
      mv.y = (mv.y == 0.0f) ? -10000.0f : mv.y;
      mv.z = (mv.z == 0.0f) ? -10000.0f : mv.z;
      mv.w = (mv.w == 0.0f) ? -10000.0f : mv.w;
      *(float4*)(S + r * kSROW + c) = mv;
    }
  }
  __syncthreads();

  // -------- phase 1: S = q_scaled K^T + mask (mask as WMMA C operand) ------
  // q was pre-scaled by 1/sqrt(32) in its projection, so a single WMMA with
  // the staged mask tile as C computes the full logits.
  {
    const int klo = hl * 8;
    v16bf afr[4];
#pragma unroll
    for (int am = 0; am < 4; am++) {
      const __bf16* qr = qp + (bh * kQ + q0 + am * 16 + l15) * kDhd;
      afr[am] = cat8(*(const v8bf*)(qr + hl * 16), *(const v8bf*)(qr + hl * 16 + 8));
    }
    const int nbase = wid * 128;  // this wave's 128 key columns
    const int dhalf = hl * 16;
    for (int nt = 0; nt < 8; nt++) {
      const int ncol = nbase + nt * 16 + l15;
      const __bf16* kr = kp + (bh * kK + ncol) * kDhd + dhalf;
      v16bf bfr = cat8(*(const v8bf*)kr, *(const v8bf*)(kr + 8));
#pragma unroll
      for (int am = 0; am < 4; am++) {
        float* sp = S + (am * 16 + hl * 8) * kSROW + ncol;
        v8f cfr;
#pragma unroll
        for (int r = 0; r < 8; r++) cfr[r] = sp[r * kSROW];  // mask tile
        v8f acc = wmma_bf16(afr[am], bfr, cfr);
#pragma unroll
        for (int r = 0; r < 8; r++) sp[r * kSROW] = acc[r];
      }
    }
  }
  __syncthreads();

  // -------- phase 2: row softmax in LDS; coalesced attn_weights store ------
  {
    const int row = tid >> 2, qt = tid & 3;
    float* srow = S + row * kSROW + qt * 256;  // this thread's quarter row
    float mx = -3.402823466e38f;
#pragma unroll 4
    for (int j = 0; j < 64; j++) {
      float4 v = *(const float4*)(srow + 4 * j);
      mx = fmaxf(mx, fmaxf(fmaxf(v.x, v.y), fmaxf(v.z, v.w)));
    }
    red[row * 4 + qt] = mx;
    __syncthreads();
    mx = fmaxf(fmaxf(red[row * 4], red[row * 4 + 1]),
               fmaxf(red[row * 4 + 2], red[row * 4 + 3]));
    float sum = 0.0f;
#pragma unroll 2
    for (int j = 0; j < 64; j++) {
      float4 v = *(const float4*)(srow + 4 * j);
      v.x = __expf(v.x - mx); v.y = __expf(v.y - mx);
      v.z = __expf(v.z - mx); v.w = __expf(v.w - mx);
      *(float4*)(srow + 4 * j) = v;
      sum += (v.x + v.y) + (v.z + v.w);
    }
    __syncthreads();
    red[row * 4 + qt] = sum;
    __syncthreads();
    if (qt == 0) {
      invr[row] = 1.0f / (red[row * 4] + red[row * 4 + 1] +
                          red[row * 4 + 2] + red[row * 4 + 3]);
    }
    __syncthreads();
    // Normalize in LDS and stream P to d_out with full-width b128 stores:
    // idx is linear over the 64x1024 tile -> 128 contiguous floats per wave.
    float* poBase = outP + (size_t)(bh * kQ + q0) * kK;
#pragma unroll 2
    for (int j = 0; j < 64; j++) {
      const int idx = (j * 256 + tid) * 4;
      const int r = idx >> 10, c = idx & 1023;
      float4 v = *(const float4*)(S + r * kSROW + c);
      const float iv = invr[r];  // wave-uniform row
      v.x *= iv; v.y *= iv; v.z *= iv; v.w *= iv;
      *(float4*)(S + r * kSROW + c) = v;
      *(float4*)(poBase + idx) = v;
    }
  }
  __syncthreads();

  // -------- phase 3: O = P @ V. P from LDS (f32->bf16). V^T tiles stream ---
  // through LDS via double-buffered TDM; DMA of tile i+1 overlaps WMMA on i.
  {
    const int dcol0 = wid * 16;  // this wave's 16 output dims
    const int klo = hl * 8;
    const int khalf = hl * 16;
    v8f z = {};
    v8f acc[4];
#pragma unroll
    for (int am = 0; am < 4; am++) acc[am] = z;

#if HAVE_TDM
    for (int it = 0; it < kK / 32; ++it) {
      if (wid == 0) {
        if (it + 1 < kK / 32) {
          tdm_issue_vtile(vpT, bh, (it + 1) * 32,
                          vstage_off + (unsigned)((it + 1) & 1) * kVBYTES);
          __builtin_amdgcn_s_wait_tensorcnt(1);  // oldest (tile it) resident
        } else {
          __builtin_amdgcn_s_wait_tensorcnt(0);
        }
      }
      __syncthreads();  // tile it visible to all waves
      v16bf bfr;
      {
        const __bf16* vr =
            vstage + (it & 1) * kVBUF + (dcol0 + l15) * kVROW + khalf;
        bfr = cat8(*(const v8bf*)vr, *(const v8bf*)(vr + 8));
      }
      const int kk = it * 32;
#pragma unroll
      for (int am = 0; am < 4; am++) {
        const float* sr = S + (am * 16 + l15) * kSROW + kk;
        v16bf afr = cvt16(*(const float4*)(sr + khalf), *(const float4*)(sr + khalf + 4),
                          *(const float4*)(sr + khalf + 8), *(const float4*)(sr + khalf + 12));
        acc[am] = wmma_bf16(afr, bfr, acc[am]);
      }
      __syncthreads();  // all waves done with tile it before it is overwritten
    }
#else
    for (int kk = 0; kk < kK; kk += 32) {
      v16bf bfr;
      {
        const __bf16* vr = vpT + (bh * kDh + dcol0 + l15) * kK + kk + khalf;
        bfr = cat8(*(const v8bf*)vr, *(const v8bf*)(vr + 8));
      }
#pragma unroll
      for (int am = 0; am < 4; am++) {
        const float* sr = S + (am * 16 + l15) * kSROW + kk;
        v16bf afr = cvt16(*(const float4*)(sr + khalf), *(const float4*)(sr + khalf + 4),
                          *(const float4*)(sr + khalf + 8), *(const float4*)(sr + khalf + 12));
        acc[am] = wmma_bf16(afr, bfr, acc[am]);
      }
    }
#endif
#pragma unroll
    for (int am = 0; am < 4; am++) {
      const int d = dcol0 + l15;
#pragma unroll
      for (int r = 0; r < 8; r++) {
        const int q = q0 + am * 16 + hl * 8 + r;
        outO[(b * kQ + q) * kE + h * kDh + d] = acc[am][r];
      }
    }
  }
}

extern "C" void kernel_launch(void* const* d_in, const int* in_sizes, int n_in,
                              void* d_out, int out_size, void* d_ws,
                              size_t ws_size, hipStream_t stream) {
  (void)in_sizes; (void)n_in; (void)out_size; (void)ws_size;
  const float* query = (const float*)d_in[0];
  const float* key   = (const float*)d_in[1];
  const float* value = (const float*)d_in[2];
  const float* mask  = (const float*)d_in[3];
  const float* qW = (const float*)d_in[4];
  const float* qb = (const float*)d_in[5];
  const float* kW = (const float*)d_in[6];
  const float* kb = (const float*)d_in[7];
  const float* vW = (const float*)d_in[8];
  const float* vb = (const float*)d_in[9];

  // Workspace: bf16 projected tensors (8 + 8 + 32 MB = 48 MB).
  __bf16* qp  = (__bf16*)d_ws;                       // [B,H,Q,32] (pre-scaled)
  __bf16* kp  = qp + (size_t)kB * kQ * kD;           // [B,H,K,32]
  __bf16* vpT = kp + (size_t)kB * kK * kD;           // [B,H,128,K] (transposed)

  float* outO = (float*)d_out;                       // [B,Q,2048]
  float* outP = outO + (size_t)kB * kQ * kE;         // [B,H,Q,K]

  const int Mtot = kB * kQ;  // 8192
  const float qscale = 0.17677669529663687f;  // 1/sqrt(32) folded into q
  proj_kernel<kD, kDhd, false>
      <<<dim3(kD / 64, Mtot / 128), 256, 0, stream>>>(query, qW, qb, qp, qscale);
  proj_kernel<kD, kDhd, false>
      <<<dim3(kD / 64, Mtot / 128), 256, 0, stream>>>(key, kW, kb, kp, 1.0f);
  proj_kernel<kE, kDh, true>
      <<<dim3(kE / 64, Mtot / 128), 256, 0, stream>>>(value, vW, vb, vpT, 1.0f);

  const size_t smem = (size_t)(64 * kSROW + 256 + 64) * sizeof(float) +
                      (size_t)2 * kVBYTES;
  attn_kernel<<<dim3(kQ / 64, kNH, kB), 256, smem, stream>>>(qp, kp, vpT, mask,
                                                             outO, outP);
}
